// sts_attention_19043884990501
// MI455X (gfx1250) — matrix operand order repaired and checked by reading, not compiled, not run on hardware
//
#include <hip/hip_runtime.h>
#include <hip/hip_bf16.h>
#include <math.h>

typedef __attribute__((ext_vector_type(16))) _Float16 v16h;
typedef __attribute__((ext_vector_type(8)))  float    v8f;

#define NB   64
#define CCH  64
#define TT   256
#define VV   25
#define TV   6400                 // T*V
#define NCTV 26214400             // N*C*T*V
#define NTILE_COLS 400            // TV/16

// ---------- K1: xm[n,c,v] = mean over T of x ----------
__global__ void mean_t_kernel(const float* __restrict__ x, float* __restrict__ xm) {
  int gid = blockIdx.x * blockDim.x + threadIdx.x;
  if (gid >= NB * CCH * VV) return;
  int v  = gid % VV;
  int nc = gid / VV;
  const float* p = x + (size_t)nc * TV + v;
  float s = 0.f;
  for (int t = 0; t < TT; ++t) s += p[t * VV];
  xm[gid] = s * (1.0f / 256.0f);
}

// ---------- K2: per (head,n) attention map att[i,n,c,v,w] ----------
__global__ void attn_kernel(const float* __restrict__ xm,
                            const float* __restrict__ WKw, const float* __restrict__ WKb,
                            const float* __restrict__ WQw, const float* __restrict__ WQb,
                            const float* __restrict__ a1w, const float* __restrict__ a1b,
                            const float* __restrict__ a2w, const float* __restrict__ a2b,
                            const float* __restrict__ fw,  const float* __restrict__ fb,
                            const float* __restrict__ alpha, const float* __restrict__ Aadj,
                            float* __restrict__ att) {
  __shared__ float xs[CCH * VV];     // 1600
  __shared__ float ks[4 * VV];       // 100
  __shared__ float qs[4 * VV];       // 100
  __shared__ float hs[8 * VV * VV];  // 5000
  const int i = blockIdx.x / NB;     // head
  const int n = blockIdx.x % NB;
  const int tid = threadIdx.x;
  for (int t = tid; t < CCH * VV; t += blockDim.x) xs[t] = xm[(size_t)n * CCH * VV + t];
  __syncthreads();
  for (int t = tid; t < 4 * VV; t += blockDim.x) {
    int r = t / VV, v = t % VV;
    float sk = WKb[i * 4 + r], sq = WQb[i * 4 + r];
    for (int c = 0; c < CCH; ++c) {
      float xv = xs[c * VV + v];
      sk += WKw[(i * 4 + r) * CCH + c] * xv;
      sq += WQw[(i * 4 + r) * CCH + c] * xv;
    }
    ks[t] = sk; qs[t] = sq;
  }
  __syncthreads();
  for (int t = tid; t < 8 * VV * VV; t += blockDim.x) {
    int o = t / (VV * VV);
    int rem = t % (VV * VV);
    int v = rem / VV, w = rem % VV;
    float s;
    if (o < 4) {
      s = a1b[i * 4 + o];
      for (int r = 0; r < 4; ++r) s += a1w[(i * 4 + o) * 4 + r] * (ks[r * VV + v] * qs[r * VV + w]);
    } else {
      int o2 = o - 4;
      s = a2b[i * 4 + o2];
      for (int r = 0; r < 4; ++r) s += a2w[(i * 4 + o2) * 4 + r] * (ks[r * VV + v] - qs[r * VV + w]);
    }
    hs[t] = tanhf(s);
  }
  __syncthreads();
  const float al = alpha[i];
  for (int t = tid; t < CCH * VV * VV; t += blockDim.x) {
    int c = t / (VV * VV);
    int rem = t % (VV * VV);
    float s = fb[i * CCH + c];
    for (int o = 0; o < 8; ++o) s += fw[(i * CCH + c) * 8 + o] * hs[o * VV * VV + rem];
    att[(((size_t)i * NB + n) * CCH + c) * (VV * VV) + rem] = s * al + Aadj[i * VV * VV + rem];
  }
}

// ---------- WMMA fragment index helpers (16x16x32 f16) ----------
__device__ __forceinline__ int a_kidx(int e, int halfsel) {
  int j = e >> 1, p = e & 1;
  return halfsel * 8 + ((j < 4) ? (2 * j) : (16 + 2 * (j - 4))) + p;
}
__device__ __forceinline__ int b_kidx(int e, int halfsel) {
  return (e < 8) ? (halfsel * 8 + e) : (16 + halfsel * 8 + (e - 8));
}

// ---------- K3/K7: Out[n,m,col] = sum_k W[m,k]*X[n,k,col] + bias[m]  (M=K=64, Ncols=6400) ----------
__global__ void gemm64_kernel(const float* __restrict__ W, const float* __restrict__ bias,
                              const float* __restrict__ X, float* __restrict__ Out) {
  const int lane = threadIdx.x & 31;
  const int wave = blockIdx.x * (blockDim.x >> 5) + (threadIdx.x >> 5);
  const int n    = wave / NTILE_COLS;
  const int ct   = wave % NTILE_COLS;
  const int col0 = ct * 16;
  const int halfsel = lane >> 4;
  const int lrow    = lane & 15;
  // prefetch first rows of this wave's B panel (emits global_prefetch_b8)
  __builtin_prefetch(X + (size_t)n * CCH * TV + col0 + lrow, 0, 1);
  __builtin_prefetch(X + ((size_t)n * CCH + 32) * TV + col0 + lrow, 0, 1);
  v8f acc[4] = {};
  for (int k0 = 0; k0 < 64; k0 += 32) {
    v16h bfrag;
#pragma unroll
    for (int e = 0; e < 16; ++e) {
      int kk = b_kidx(e, halfsel);
      bfrag[e] = (_Float16)X[((size_t)n * CCH + k0 + kk) * TV + col0 + lrow];
    }
#pragma unroll
    for (int rb = 0; rb < 4; ++rb) {
      v16h afrag;
#pragma unroll
      for (int e = 0; e < 16; ++e) {
        int kk = a_kidx(e, halfsel);
        afrag[e] = (_Float16)W[(rb * 16 + lrow) * 64 + k0 + kk];
      }
      acc[rb] = __builtin_amdgcn_wmma_f32_16x16x32_f16(false, afrag, false, bfrag,
                                                       (short)0, acc[rb], false, false);
    }
  }
#pragma unroll
  for (int rb = 0; rb < 4; ++rb) {
#pragma unroll
    for (int r = 0; r < 8; ++r) {
      int m = rb * 16 + r + halfsel * 8;
      Out[((size_t)n * CCH + m) * TV + col0 + lrow] = acc[rb][r] + bias[m];
    }
  }
}

// ---------- K4: out[n,c,t,v] += sum_w att[n,c,v,w] * vbuf[n,c,t,w]  (WMMA, K=25 padded to 32) ----------
__global__ void att_apply_kernel(const float* __restrict__ att,  // [N,64,625] (this head)
                                 const float* __restrict__ vbuf, // [N,64,256,25]
                                 float* __restrict__ out) {
  const int lane = threadIdx.x & 31;
  const int wave = blockIdx.x * (blockDim.x >> 5) + (threadIdx.x >> 5);
  const int n   = wave >> 10;          // 64*16 waves per n
  const int rem = wave & 1023;
  const int c   = rem >> 4;
  const int t0  = (rem & 15) << 4;
  const int halfsel = lane >> 4;
  const int lrow    = lane & 15;
  const float* vb = vbuf + ((size_t)n * CCH + c) * TV;
  const float* at = att  + ((size_t)n * CCH + c) * (VV * VV);
  float* ob = out + ((size_t)n * CCH + c) * TV;
  // A = v-tile [16 x 32], rows = t, cols = w (zero pad w>=25)
  v16h afrag;
#pragma unroll
  for (int e = 0; e < 16; ++e) {
    int kk = a_kidx(e, halfsel);
    afrag[e] = (kk < VV) ? (_Float16)vb[(t0 + lrow) * VV + kk] : (_Float16)0.0f;
  }
#pragma unroll
  for (int nt = 0; nt < 2; ++nt) {
    const int v = nt * 16 + lrow;  // output joint column
    v16h bfrag;
#pragma unroll
    for (int e = 0; e < 16; ++e) {
      int kk = b_kidx(e, halfsel);
      bfrag[e] = (kk < VV && v < VV) ? (_Float16)at[v * VV + kk] : (_Float16)0.0f;
    }
    v8f cacc;
#pragma unroll
    for (int r = 0; r < 8; ++r) {
      int t = t0 + r + halfsel * 8;
      cacc[r] = (v < VV) ? ob[t * VV + v] : 0.0f;
    }
    cacc = __builtin_amdgcn_wmma_f32_16x16x32_f16(false, afrag, false, bfrag,
                                                  (short)0, cacc, false, false);
    if (v < VV) {
#pragma unroll
      for (int r = 0; r < 8; ++r) {
        int t = t0 + r + halfsel * 8;
        ob[t * VV + v] = cacc[r];
      }
    }
  }
}

// ---------- BN stats (training mode, stats over N,T,V): one block per channel ----------
__global__ void bnstats_kernel(const float* __restrict__ data, int nch,
                               const float* __restrict__ g, const float* __restrict__ b,
                               float* __restrict__ scale, float* __restrict__ shift) {
  __shared__ float ssum[256], ssq[256];
  const int c = blockIdx.x;
  const int tid = threadIdx.x;
  float s = 0.f, q = 0.f;
  for (int j = tid; j < NB * TV; j += blockDim.x) {
    int n = j / TV;
    int r = j - n * TV;
    float v = data[((size_t)n * nch + c) * TV + r];
    s += v; q += v * v;
  }
  ssum[tid] = s; ssq[tid] = q;
  __syncthreads();
  for (int st = 128; st > 0; st >>= 1) {
    if (tid < st) { ssum[tid] += ssum[tid + st]; ssq[tid] += ssq[tid + st]; }
    __syncthreads();
  }
  if (tid == 0) {
    const float cnt = (float)(NB * TV);
    float mean = ssum[0] / cnt;
    float var  = ssq[0] / cnt - mean * mean;
    float rstd = rsqrtf(var + 1e-5f);
    float sc = g[c] * rstd;
    scale[c] = sc;
    shift[c] = b[c] - mean * sc;
  }
}

// ---------- y = relu(bn(out) + x) ----------
__global__ void bn_res_relu_kernel(const float* __restrict__ out, const float* __restrict__ x,
                                   const float* __restrict__ scale, const float* __restrict__ shift,
                                   float* __restrict__ y) {
  size_t idx = (size_t)blockIdx.x * blockDim.x + threadIdx.x;
  if (idx >= (size_t)NCTV) return;
  int c = (int)((idx / TV) & 63);
  float v = out[idx] * scale[c] + shift[c] + x[idx];
  y[idx] = v > 0.f ? v : 0.f;
}

// ---------- gather stacked branch weights ----------
__global__ void gather_kernel(const float* b1w, const float* b1b, const float* b1g1, const float* b1be1,
                              const float* b2w, const float* b2b, const float* b2g1, const float* b2be1,
                              const float* b3w, const float* b3b, const float* b3g1, const float* b3be1,
                              const float* b4w, const float* b4b, const float* b4g1, const float* b4be1,
                              const float* b1g2, const float* b1be2,
                              const float* b2g2, const float* b2be2,
                              const float* b3g2, const float* b3be2,
                              float* Wst, float* Bst, float* G1, float* Be1, float* G2, float* Be2) {
  int tid = blockIdx.x * blockDim.x + threadIdx.x;
  const float* ws_[4]  = {b1w, b2w, b3w, b4w};
  const float* bs_[4]  = {b1b, b2b, b3b, b4b};
  const float* g1_[4]  = {b1g1, b2g1, b3g1, b4g1};
  const float* be1_[4] = {b1be1, b2be1, b3be1, b4be1};
  const float* g2_[3]  = {b1g2, b2g2, b3g2};
  const float* be2_[3] = {b1be2, b2be2, b3be2};
  if (tid < 4096) Wst[tid] = ws_[tid / 1024][tid % 1024];
  if (tid < 64) {
    Bst[tid] = bs_[tid / 16][tid % 16];
    G1[tid]  = g1_[tid / 16][tid % 16];
    Be1[tid] = be1_[tid / 16][tid % 16];
  }
  if (tid < 48) {
    G2[tid]  = g2_[tid / 16][tid % 16];
    Be2[tid] = be2_[tid / 16][tid % 16];
  }
}

// ---------- apply first branch BN (+relu for branches 1-3) ----------
__global__ void bn1_apply_kernel(const float* __restrict__ z,
                                 const float* __restrict__ scale, const float* __restrict__ shift,
                                 float* __restrict__ a) {
  size_t idx = (size_t)blockIdx.x * blockDim.x + threadIdx.x;
  if (idx >= (size_t)NCTV) return;
  int ch = (int)((idx / TV) & 63);
  float v = z[idx] * scale[ch] + shift[ch];
  if (ch < 48) v = v > 0.f ? v : 0.f;
  a[idx] = v;
}

// ---------- temporal 5x3 convs (branches 1,2) as implicit-im2col WMMA GEMM ----------
// out[n, br, o(16), col] = bias[o] + sum_{k<240} W[o,k] * im2col[k, col]
// k = ic*15 + kt*3 + kv  (matches [o, ic, kt, kv] weight layout), K padded to 256.
__global__ void tconv_wmma_kernel(const float* __restrict__ a,
                                  const float* __restrict__ w1, const float* __restrict__ b1f,
                                  const float* __restrict__ w2, const float* __restrict__ b2f,
                                  float* __restrict__ cbuf) {
  const int lane = threadIdx.x & 31;
  const int wave = blockIdx.x * (blockDim.x >> 5) + (threadIdx.x >> 5);
  // waves: (n*2+br) x 400 column tiles = 51200
  const int ct   = wave % NTILE_COLS;
  const int nb2  = wave / NTILE_COLS;
  const int br   = nb2 & 1;
  const int n    = nb2 >> 1;
  const int col0 = ct * 16;
  const int halfsel = lane >> 4;
  const int lrow    = lane & 15;
  const float* w  = br ? w2 : w1;
  const float* bf = br ? b2f : b1f;
  const int padT = br ? 4 : 2, padV = br ? 2 : 1, d = br ? 2 : 1;
  const float* ain = a + ((size_t)n * CCH + br * 16) * TV;
  const int col = col0 + lrow;            // output/B column
  const int t = col / VV, v = col % VV;
  __builtin_prefetch(ain + col, 0, 1);    // global_prefetch_b8
  v8f acc = {};
  for (int k0 = 0; k0 < 256; k0 += 32) {
    v16h afrag, bfrag;
#pragma unroll
    for (int e = 0; e < 16; ++e) {
      int ka = k0 + a_kidx(e, halfsel);
      afrag[e] = (ka < 240) ? (_Float16)w[lrow * 240 + ka] : (_Float16)0.0f;
      int kb = k0 + b_kidx(e, halfsel);
      float val = 0.f;
      if (kb < 240) {
        int ic = kb / 15;
        int rr = kb - ic * 15;
        int kt = rr / 3;
        int kv = rr - kt * 3;
        int ti = t - padT + kt * d;
        int vi = v - padV + kv * d;
        if (ti >= 0 && ti < TT && vi >= 0 && vi < VV)
          val = ain[(size_t)ic * TV + ti * VV + vi];
      }
      bfrag[e] = (_Float16)val;
    }
    acc = __builtin_amdgcn_wmma_f32_16x16x32_f16(false, afrag, false, bfrag,
                                                 (short)0, acc, false, false);
  }
#pragma unroll
  for (int r = 0; r < 8; ++r) {
    int o = r + halfsel * 8;
    cbuf[((size_t)n * 48 + br * 16 + o) * TV + col] = acc[r] + bf[o];
  }
}

// ---------- maxpool (3,1) over T with -inf pad, branch 3 ----------
__global__ void maxpool_kernel(const float* __restrict__ a, float* __restrict__ cbuf) {
  size_t idx = (size_t)blockIdx.x * blockDim.x + threadIdx.x;
  if (idx >= (size_t)NB * 16 * TV) return;
  int v = (int)(idx % VV);
  int t = (int)((idx / VV) & 255);
  int o = (int)((idx / TV) & 15);
  int n = (int)(idx / ((size_t)16 * TV));
  const float* ap = a + ((size_t)n * CCH + 32 + o) * TV + v;
  float m = -INFINITY;
  for (int dt = -1; dt <= 1; ++dt) {
    int ti = t + dt;
    if (ti >= 0 && ti < TT) { float vv = ap[ti * VV]; m = vv > m ? vv : m; }
  }
  cbuf[(((size_t)n * 48 + 32 + o) * TT + t) * VV + v] = m;
}

// ---------- final: relu(concat(bn2(c1..c3), o4) + x) ----------
__global__ void final_kernel(const float* __restrict__ cbuf, const float* __restrict__ abuf,
                             const float* __restrict__ x,
                             const float* __restrict__ scaleF, const float* __restrict__ shiftF,
                             float* __restrict__ out) {
  size_t idx = (size_t)blockIdx.x * blockDim.x + threadIdx.x;
  if (idx >= (size_t)NCTV) return;
  int r  = (int)(idx % TV);
  int ch = (int)((idx / TV) & 63);
  int n  = (int)(idx / ((size_t)CCH * TV));
  float val;
  if (ch < 48) val = cbuf[((size_t)n * 48 + ch) * TV + r] * scaleF[ch] + shiftF[ch];
  else         val = abuf[idx];
  float s = val + x[idx];
  out[idx] = s > 0.f ? s : 0.f;
}

extern "C" void kernel_launch(void* const* d_in, const int* in_sizes, int n_in,
                              void* d_out, int out_size, void* d_ws, size_t ws_size,
                              hipStream_t stream) {
  const float* x     = (const float*)d_in[0];
  const float* WKw   = (const float*)d_in[1];
  const float* WKb   = (const float*)d_in[2];
  const float* WQw   = (const float*)d_in[3];
  const float* WQb   = (const float*)d_in[4];
  const float* WVw   = (const float*)d_in[5];
  const float* WVb   = (const float*)d_in[6];
  const float* a1w   = (const float*)d_in[7];
  const float* a1b   = (const float*)d_in[8];
  const float* a2w   = (const float*)d_in[9];
  const float* a2b   = (const float*)d_in[10];
  const float* fw    = (const float*)d_in[11];
  const float* fb    = (const float*)d_in[12];
  const float* alpha = (const float*)d_in[13];
  const float* Aadj  = (const float*)d_in[14];
  const float* bng   = (const float*)d_in[15];
  const float* bnb   = (const float*)d_in[16];
  const float* b1w = (const float*)d_in[17], *b1b = (const float*)d_in[18];
  const float* b1g1 = (const float*)d_in[19], *b1be1 = (const float*)d_in[20];
  const float* b1fw = (const float*)d_in[21], *b1fb = (const float*)d_in[22];
  const float* b1g2 = (const float*)d_in[23], *b1be2 = (const float*)d_in[24];
  const float* b2w = (const float*)d_in[25], *b2b = (const float*)d_in[26];
  const float* b2g1 = (const float*)d_in[27], *b2be1 = (const float*)d_in[28];
  const float* b2fw = (const float*)d_in[29], *b2fb = (const float*)d_in[30];
  const float* b2g2 = (const float*)d_in[31], *b2be2 = (const float*)d_in[32];
  const float* b3w = (const float*)d_in[33], *b3b = (const float*)d_in[34];
  const float* b3g1 = (const float*)d_in[35], *b3be1 = (const float*)d_in[36];
  const float* b3g2 = (const float*)d_in[37], *b3be2 = (const float*)d_in[38];
  const float* b4w = (const float*)d_in[39], *b4b = (const float*)d_in[40];
  const float* b4g1 = (const float*)d_in[41], *b4be1 = (const float*)d_in[42];

  float* ws = (float*)d_ws;
  float* BUF0 = ws;                        // attention out  -> later stacked 1x1 outputs z
  float* BUF1 = ws + (size_t)NCTV;         // v (per head)   -> later post-BN1 activations a
  float* BUF2 = ws + (size_t)2 * NCTV;     // y              -> later conv/pool outputs c (48ch)
  float* ATT  = ws + (size_t)3 * NCTV;     // 3*64*64*625 = 7,680,000
  float* XM   = ATT + 7680000;             // 102,400
  float* SC1 = XM + 102400;  float* SH1 = SC1 + 64;
  float* SC2 = SH1 + 64;     float* SH2 = SC2 + 64;
  float* SCF = SH2 + 64;     float* SHF = SCF + 48;
  float* WST = SHF + 48;     float* BST = WST + 4096;
  float* G1  = BST + 64;     float* BE1 = G1 + 64;
  float* G2  = BE1 + 64;     float* BE2 = G2 + 48;

  const int EW_BLOCKS = NCTV / 256;  // 102400

  // 1) mean over T
  mean_t_kernel<<<400, 256, 0, stream>>>(x, XM);
  // 2) attention maps for all 3 heads
  attn_kernel<<<192, 256, 0, stream>>>(XM, WKw, WKb, WQw, WQb, a1w, a1b, a2w, a2b,
                                       fw, fb, alpha, Aadj, ATT);
  // 3) accumulate heads: out = sum_i att_i @ (W_i x + b_i)
  hipMemsetAsync(BUF0, 0, (size_t)NCTV * sizeof(float), stream);
  for (int i = 0; i < 3; ++i) {
    gemm64_kernel<<<3200, 256, 0, stream>>>(WVw + (size_t)i * 4096, WVb + (size_t)i * 64, x, BUF1);
    att_apply_kernel<<<8192, 256, 0, stream>>>(ATT + (size_t)i * NB * CCH * VV * VV, BUF1, BUF0);
  }
  // 4) y = relu(bn(out) + x)
  bnstats_kernel<<<64, 256, 0, stream>>>(BUF0, 64, bng, bnb, SC1, SH1);
  bn_res_relu_kernel<<<EW_BLOCKS, 256, 0, stream>>>(BUF0, x, SC1, SH1, BUF2);
  // 5) stack branch 1x1 weights, then one WMMA GEMM for all 4 branches
  gather_kernel<<<16, 256, 0, stream>>>(b1w, b1b, b1g1, b1be1, b2w, b2b, b2g1, b2be1,
                                        b3w, b3b, b3g1, b3be1, b4w, b4b, b4g1, b4be1,
                                        b1g2, b1be2, b2g2, b2be2, b3g2, b3be2,
                                        WST, BST, G1, BE1, G2, BE2);
  gemm64_kernel<<<3200, 256, 0, stream>>>(WST, BST, BUF2, BUF0);
  // 6) first BN per branch (+relu for branches 1-3)
  bnstats_kernel<<<64, 256, 0, stream>>>(BUF0, 64, G1, BE1, SC2, SH2);
  bn1_apply_kernel<<<EW_BLOCKS, 256, 0, stream>>>(BUF0, SC2, SH2, BUF1);
  // 7) temporal convs (branches 1,2) via WMMA im2col, maxpool (branch 3)
  tconv_wmma_kernel<<<6400, 256, 0, stream>>>(BUF1, b1fw, b1fb, b2fw, b2fb, BUF2);
  maxpool_kernel<<<25600, 256, 0, stream>>>(BUF1, BUF2);
  // 8) second BN (48 channels), final concat + residual + relu
  bnstats_kernel<<<48, 256, 0, stream>>>(BUF2, 48, G2, BE2, SCF, SHF);
  final_kernel<<<EW_BLOCKS, 256, 0, stream>>>(BUF2, BUF1, x, SCF, SHF, (float*)d_out);
}
